// CascadedAttentionCell_78048145703167
// MI455X (gfx1250) — compile-verified, matrix operand order checked by reading, not run on hardware
//
#include <hip/hip_runtime.h>
#include <cstdint>

#define BB 128
#define SSTEPS 96
#define TT 256
#define DD 1024
#define VV 28

typedef __attribute__((ext_vector_type(16))) __bf16 v16bf;
typedef __attribute__((ext_vector_type(8)))  __bf16 v8bf;
typedef __attribute__((ext_vector_type(8)))  float  v8f;

union FragAB { v16bf v; v8bf h[2]; };

__device__ __forceinline__ float bf2f(uint32_t lo16) {
    return __uint_as_float(lo16 << 16);
}
__device__ __forceinline__ uint16_t f2bf(float f) {
    uint32_t x = __float_as_uint(f);
    x += 0x7FFFu + ((x >> 16) & 1u);   // round-nearest-even
    return (uint16_t)(x >> 16);
}
// cheap tanh: 2 transcendentals + ~5 VALU (vs ~20+ for libm tanhf)
__device__ __forceinline__ float fast_tanh(float x) {
    x = fminf(fmaxf(x, -10.f), 10.f);
    float e = __expf(2.f * x);
    return __fdividef(e - 1.f, e + 1.f);
}

// ---------------- pack kernels ----------------
__global__ void k_pack_x(const float* __restrict__ x, uint16_t* __restrict__ xbf, int n4) {
    int i = blockIdx.x * blockDim.x + threadIdx.x;
    if (i >= n4) return;
    float4 v = ((const float4*)x)[i];
    uint2 o;
    o.x = (uint32_t)f2bf(v.x) | ((uint32_t)f2bf(v.y) << 16);
    o.y = (uint32_t)f2bf(v.z) | ((uint32_t)f2bf(v.w) << 16);
    ((uint2*)xbf)[i] = o;
}

__global__ void k_pack_uat(const float* __restrict__ Ua, uint16_t* __restrict__ uat) {
    int i = blockIdx.x * 256 + threadIdx.x;   // i in [0, D*D)
    int n = i >> 10, k = i & (DD - 1);
    uat[i] = f2bf(Ua[k * DD + n]);            // UaT[n,k] = Ua[k,n]
}

__global__ void k_copy(const float* __restrict__ src, float* __restrict__ dst, int n) {
    int i = blockIdx.x * blockDim.x + threadIdx.x;
    if (i < n) dst[i] = src[i];
}

// ---------------- WMMA GEMM: UaH = x @ Ua (bf16 in, f32 acc, bf16 out) ----------------
// Software-pipelined: fragments for K-step kk+32 are loaded before the WMMAs
// consuming K-step kk, so v_wmma waits on partial loadcnt instead of a full drain.
__global__ void __launch_bounds__(256)
k_gemm_uah(const uint16_t* __restrict__ xbf, const uint16_t* __restrict__ uat,
           uint16_t* __restrict__ uah) {
    const int lane  = threadIdx.x & 31;
    const int wv    = threadIdx.x >> 5;
    const int m0    = blockIdx.x * 16;              // M tile (B*T rows)
    const int n0    = blockIdx.y * 512 + wv * 64;   // 4 N tiles per wave
    const int r16   = lane & 15;
    const int khalf = (lane >> 4) << 3;             // lanes>=16 hold K+8 halves

    const size_t arow = (size_t)(m0 + r16) * DD;
    size_t brow[4];
    #pragma unroll
    for (int j = 0; j < 4; ++j) brow[j] = (size_t)(n0 + j * 16 + r16) * DD;

    v8f z = {0.f,0.f,0.f,0.f,0.f,0.f,0.f,0.f};
    v8f acc[4] = {z, z, z, z};

    FragAB A[2], Bf[2][4];
    auto loadk = [&](int buf, int ko) {
        A[buf].h[0] = *(const v8bf*)(xbf + arow + ko);
        A[buf].h[1] = *(const v8bf*)(xbf + arow + ko + 16);
        #pragma unroll
        for (int j = 0; j < 4; ++j) {
            Bf[buf][j].h[0] = *(const v8bf*)(uat + brow[j] + ko);
            Bf[buf][j].h[1] = *(const v8bf*)(uat + brow[j] + ko + 16);
        }
    };

    loadk(0, khalf);
    int cur = 0;
    for (int kk = 0; kk < DD; kk += 32) {
        const int nxt = cur ^ 1;
        if (kk + 32 < DD) loadk(nxt, kk + 32 + khalf);   // prefetch next K-step
        #pragma unroll
        for (int j = 0; j < 4; ++j)
            acc[j] = __builtin_amdgcn_wmma_f32_16x16x32_bf16(
                false, A[cur].v, false, Bf[cur][j].v, (short)0, acc[j], false, false);
        cur = nxt;
    }

    const int rowadd = (lane >> 4) << 3;   // C layout: lanes>=16 hold rows r+8
    #pragma unroll
    for (int r = 0; r < 8; ++r) {
        size_t row = (size_t)(m0 + r + rowadd) * DD;
        #pragma unroll
        for (int j = 0; j < 4; ++j)
            uah[row + n0 + j * 16 + r16] = f2bf(acc[j][r]);
    }
}

// ---------------- scan step kernels ----------------
// WaS[b,d] = sum_v y_prev[b,v] * Wa[v,d]
__global__ void __launch_bounds__(256)
k_was(const float* __restrict__ yprev, const float* __restrict__ Wa, float* __restrict__ was) {
    __shared__ float sy[VV];
    const int b = blockIdx.x >> 2;
    const int d = ((blockIdx.x & 3) << 8) + threadIdx.x;
    if (threadIdx.x < VV) sy[threadIdx.x] = yprev[b * VV + threadIdx.x];
    __syncthreads();
    float acc = 0.f;
    #pragma unroll
    for (int v = 0; v < VV; ++v) acc += sy[v] * Wa[v * DD + d];
    was[(size_t)b * DD + d] = acc;
}

// scores[b,t] = sum_d tanh(UaH[b,t,d] + WaS[b,d]) * va[d]   (one wave per (b,t))
__global__ void __launch_bounds__(256)
k_scores(const uint16_t* __restrict__ uah, const float* __restrict__ was,
         const float* __restrict__ va, float* __restrict__ scores) {
    const int lane = threadIdx.x & 31;
    const int bt   = blockIdx.x * 8 + (threadIdx.x >> 5);
    const int b    = bt >> 8;   // T == 256
    const uint16_t* up = uah + (size_t)bt * DD;
    const float*    wp = was + (size_t)b * DD;
    float acc = 0.f;
    #pragma unroll
    for (int c = 0; c < 4; ++c) {
        const int d0 = c * 256 + lane * 8;
        uint4  u  = *(const uint4*)(up + d0);
        float4 w0 = *(const float4*)(wp + d0);
        float4 w1 = *(const float4*)(wp + d0 + 4);
        float4 a0 = *(const float4*)(va + d0);
        float4 a1 = *(const float4*)(va + d0 + 4);
        acc += fast_tanh(bf2f(u.x & 0xFFFFu) + w0.x) * a0.x;
        acc += fast_tanh(bf2f(u.x >> 16)     + w0.y) * a0.y;
        acc += fast_tanh(bf2f(u.y & 0xFFFFu) + w0.z) * a0.z;
        acc += fast_tanh(bf2f(u.y >> 16)     + w0.w) * a0.w;
        acc += fast_tanh(bf2f(u.z & 0xFFFFu) + w1.x) * a1.x;
        acc += fast_tanh(bf2f(u.z >> 16)     + w1.y) * a1.y;
        acc += fast_tanh(bf2f(u.w & 0xFFFFu) + w1.z) * a1.z;
        acc += fast_tanh(bf2f(u.w >> 16)     + w1.w) * a1.w;
    }
    #pragma unroll
    for (int off = 16; off > 0; off >>= 1) acc += __shfl_xor(acc, off, 32);
    if (lane == 0) scores[bt] = acc;
}

// in-place softmax over T per batch row
__global__ void __launch_bounds__(256)
k_softmax(float* __restrict__ scores) {
    __shared__ float sd[TT];
    const int b = blockIdx.x, t = threadIdx.x;
    float s = scores[b * TT + t];
    sd[t] = s; __syncthreads();
    for (int off = 128; off > 0; off >>= 1) {
        if (t < off) sd[t] = fmaxf(sd[t], sd[t + off]);
        __syncthreads();
    }
    float mx = sd[0]; __syncthreads();
    float e = __expf(s - mx);
    sd[t] = e; __syncthreads();
    for (int off = 128; off > 0; off >>= 1) {
        if (t < off) sd[t] += sd[t + off];
        __syncthreads();
    }
    scores[b * TT + t] = e / sd[0];
}

// ctx[b,d] = sum_t sm[b,t] * x[b,t,d]   (x in bf16)
__global__ void __launch_bounds__(128)
k_ctx(const float* __restrict__ sm, const uint16_t* __restrict__ xbf, float* __restrict__ ctx) {
    __shared__ float ssm[TT];
    const int b = blockIdx.x, tid = threadIdx.x;
    ssm[tid]       = sm[b * TT + tid];
    ssm[tid + 128] = sm[b * TT + tid + 128];
    __syncthreads();
    float acc[8] = {0.f,0.f,0.f,0.f,0.f,0.f,0.f,0.f};
    const uint16_t* xp = xbf + (size_t)b * TT * DD + tid * 8;
    #pragma unroll 4
    for (int t = 0; t < TT; ++t) {
        const float s = ssm[t];
        uint4 u = *(const uint4*)(xp + (size_t)t * DD);
        acc[0] += s * bf2f(u.x & 0xFFFFu);
        acc[1] += s * bf2f(u.x >> 16);
        acc[2] += s * bf2f(u.y & 0xFFFFu);
        acc[3] += s * bf2f(u.y >> 16);
        acc[4] += s * bf2f(u.z & 0xFFFFu);
        acc[5] += s * bf2f(u.z >> 16);
        acc[6] += s * bf2f(u.w & 0xFFFFu);
        acc[7] += s * bf2f(u.w >> 16);
    }
    float* cp = ctx + (size_t)b * DD + tid * 8;
    #pragma unroll
    for (int i = 0; i < 8; ++i) cp[i] = acc[i];
}

// y = sigmoid(Emb[argmax(y_prev)] @ Wo + h @ Uo + ctx @ Co)
__global__ void __launch_bounds__(256)
k_out(const float* __restrict__ yprev, const float* __restrict__ inputs,
      const float* __restrict__ ctx, const float* __restrict__ Uo,
      const float* __restrict__ Co, const float* __restrict__ Emb,
      const float* __restrict__ Wo, float* __restrict__ ynext,
      float* __restrict__ out, int step) {
    __shared__ float lsum[8][VV];
    __shared__ int s_am;
    const int b = blockIdx.x, tid = threadIdx.x;
    if (tid == 0) {
        int am = 0; float best = yprev[b * VV];
        for (int v = 1; v < VV; ++v) {
            float y = yprev[b * VV + v];
            if (y > best) { best = y; am = v; }   // first max, like jnp.argmax
        }
        s_am = am;
    }
    float acc[VV];
    #pragma unroll
    for (int v = 0; v < VV; ++v) acc[v] = 0.f;
    const float* hp = inputs + ((size_t)b * SSTEPS + step) * DD;
    const float* cp = ctx + (size_t)b * DD;
    for (int d = tid; d < DD; d += 256) {
        const float h = hp[d], c = cp[d];
        const float* uo = Uo + d * VV;
        const float* co = Co + d * VV;
        #pragma unroll
        for (int v = 0; v < VV; ++v) acc[v] += h * uo[v] + c * co[v];
    }
    const int lane = tid & 31, wv = tid >> 5;
    #pragma unroll
    for (int v = 0; v < VV; ++v) {
        float a = acc[v];
        #pragma unroll
        for (int off = 16; off > 0; off >>= 1) a += __shfl_xor(a, off, 32);
        if (lane == 0) lsum[wv][v] = a;
    }
    __syncthreads();
    if (tid < VV) {
        const int v = tid;
        float s = 0.f;
        #pragma unroll
        for (int w = 0; w < 8; ++w) s += lsum[w][v];
        const int am = s_am;
        float e = 0.f;
        #pragma unroll
        for (int u = 0; u < VV; ++u) e += Emb[am * VV + u] * Wo[u * VV + v];
        const float y = 1.f / (1.f + __expf(-(s + e)));
        ynext[b * VV + v] = y;
        out[((size_t)b * SSTEPS + step) * VV + v] = y;
    }
}

// ---------------- host launcher ----------------
extern "C" void kernel_launch(void* const* d_in, const int* in_sizes, int n_in,
                              void* d_out, int out_size, void* d_ws, size_t ws_size,
                              hipStream_t stream) {
    const float* inputs = (const float*)d_in[0];
    const float* x      = (const float*)d_in[1];
    const float* y0     = (const float*)d_in[2];
    const float* Wa     = (const float*)d_in[3];
    const float* Ua     = (const float*)d_in[4];
    const float* Va     = (const float*)d_in[5];
    const float* Wo     = (const float*)d_in[6];
    const float* Uo     = (const float*)d_in[7];
    const float* Co     = (const float*)d_in[8];
    const float* Emb    = (const float*)d_in[9];
    float* out = (float*)d_out;

    char* ws = (char*)d_ws;
    uint16_t* xbf    = (uint16_t*)(ws);                               // 64 MiB
    uint16_t* uat    = (uint16_t*)(ws + 67108864);                    //  2 MiB
    uint16_t* uah    = (uint16_t*)(ws + 69206016);                    // 64 MiB
    float*    was    = (float*)   (ws + 136314880);                   // 512 KiB
    float*    scores = (float*)   (ws + 136839168);                   // 128 KiB
    float*    ctx    = (float*)   (ws + 136970240);                   // 512 KiB
    float*    yA     = (float*)   (ws + 137494528);                   // 14 KiB
    float*    yB     = yA + BB * VV;

    // Stage bf16 copies (x and Ua^T), then one big WMMA GEMM for UaH.
    const int n4 = BB * TT * DD / 4;
    k_pack_x  <<<(n4 + 255) / 256, 256, 0, stream>>>(x, xbf, n4);
    k_pack_uat<<<DD * DD / 256,   256, 0, stream>>>(Ua, uat);
    dim3 ggrid(BB * TT / 16, DD / 512);
    k_gemm_uah<<<ggrid, 256, 0, stream>>>(xbf, uat, uah);
    k_copy    <<<(BB * VV + 255) / 256, 256, 0, stream>>>(y0, yA, BB * VV);

    for (int s = 0; s < SSTEPS; ++s) {
        float* yp = (s & 1) ? yB : yA;
        float* yn = (s & 1) ? yA : yB;
        k_was    <<<BB * 4,      256, 0, stream>>>(yp, Wa, was);
        k_scores <<<BB * TT / 8, 256, 0, stream>>>(uah, was, Va, scores);
        k_softmax<<<BB,          256, 0, stream>>>(scores);
        k_ctx    <<<BB,          128, 0, stream>>>(scores, xbf, ctx);
        k_out    <<<BB,          256, 0, stream>>>(yp, inputs, ctx, Uo, Co, Emb, Wo, yn, out, s);
    }
}
